// ComplexAttention_71219147702408
// MI455X (gfx1250) — matrix operand order repaired and checked by reading, not compiled, not run on hardware
//
#include <hip/hip_runtime.h>

// Problem constants (match reference)
#define BB 4
#define SS 1024
#define DD 1024
#define HH 16
#define HD 64
#define BSD (BB * SS * DD)  // 4,194,304

typedef __attribute__((ext_vector_type(16))) __bf16 bf16x16;
typedef __attribute__((ext_vector_type(8))) float f32x8;
typedef __attribute__((ext_vector_type(4))) unsigned int u32x4;
typedef __attribute__((ext_vector_type(4))) int i32x4;
typedef __attribute__((ext_vector_type(8))) int i32x8;

union Tile {
  bf16x16 v;
  u32x4 q[2];
  unsigned int u[8];
  unsigned short h[16];
};

// D = A*B + C
__device__ __forceinline__ f32x8 wmma_bf16(const Tile& a, const Tile& b, f32x8 c) {
  return __builtin_amdgcn_wmma_f32_16x16x32_bf16(false, a.v, false, b.v,
                                                 (short)0, c, false, false);
}
// D = A*B - C   (CNeg: lowers to neg_lo:[0,0,1], verified in round 2)
__device__ __forceinline__ f32x8 wmma_bf16_cn(const Tile& a, const Tile& b, f32x8 c) {
  return __builtin_amdgcn_wmma_f32_16x16x32_bf16(false, a.v, false, b.v,
                                                 (short)1, c, false, false);
}

__device__ __forceinline__ unsigned short f2bf(float f) {
  unsigned int u = __float_as_uint(f);
  u += 0x7FFFu + ((u >> 16) & 1u);  // round-to-nearest-even
  return (unsigned short)(u >> 16);
}

// ---------------------------------------------------------------- TDM helper
#if __has_builtin(__builtin_amdgcn_tensor_load_to_lds)
#define HAVE_TDM 1
#endif

// 2-D tile load Global->LDS via Tensor Data Mover.
// Tile: tile_d1 rows x tile_d0 elems (bf16), row stride = stride0 elems.
// LDS result: compact [tile_d1][tile_d0] bf16 at lds byte offset lds_addr.
__device__ __forceinline__ void tdm_load_2d(unsigned int lds_addr,
                                            const unsigned short* gptr,
                                            unsigned int tensor_d0,
                                            unsigned int tensor_d1,
                                            unsigned int tile_d0,
                                            unsigned int tile_d1,
                                            unsigned long long stride0) {
#ifdef HAVE_TDM
  const unsigned long long ga = (unsigned long long)(uintptr_t)gptr;
  u32x4 g0;
  g0[0] = 1u;                                   // count=1, user descriptor
  g0[1] = lds_addr;                             // lds_addr [63:32]
  g0[2] = (unsigned int)(ga & 0xFFFFFFFFu);     // global_addr lo [95:64]
  g0[3] = (unsigned int)((ga >> 32) & 0x1FFFFFFu) | (2u << 30);  // hi + type=2
  i32x8 g1;
  g1[0] = (int)(1u << 16);                      // data_size=1 (2 bytes)
  g1[1] = (int)((tensor_d0 & 0xFFFFu) << 16);   // tensor_dim0 lo16 @bit48
  g1[2] = (int)(((tensor_d0 >> 16) & 0xFFFFu) | ((tensor_d1 & 0xFFFFu) << 16));
  g1[3] = (int)(((tensor_d1 >> 16) & 0xFFFFu) | ((tile_d0 & 0xFFFFu) << 16));
  g1[4] = (int)(tile_d1 & 0xFFFFu);             // tile_dim1; tile_dim2=0
  g1[5] = (int)(unsigned int)(stride0 & 0xFFFFFFFFull);  // dim0_stride lo32
  g1[6] = (int)(unsigned int)((stride0 >> 32) & 0xFFFFull);  // hi16; dim1_stride=0
  g1[7] = 0;
  const i32x4 gz = {0, 0, 0, 0};
#if defined(__clang_major__) && (__clang_major__ >= 23)
  const i32x8 gz8 = {0, 0, 0, 0, 0, 0, 0, 0};
  __builtin_amdgcn_tensor_load_to_lds(g0, g1, gz, gz, gz8, 0);
#else
  __builtin_amdgcn_tensor_load_to_lds(g0, g1, gz, gz, 0);
#endif
#endif
}

__device__ __forceinline__ unsigned int lds_byte_off(const void* p) {
  // generic address of an LDS object: low 32 bits are the LDS byte offset
  return (unsigned int)(uintptr_t)p;
}

// ---------------------------------------------------------------- convert
__global__ void cvt_f32_bf16(const float* __restrict__ in,
                             unsigned short* __restrict__ out, int n) {
  for (int i = blockIdx.x * blockDim.x + threadIdx.x; i < n;
       i += gridDim.x * blockDim.x)
    out[i] = f2bf(in[i]);
}

// ---------------------------------------------------------------- complex GEMM
// C = X @ W^T (complex).  X:[M,K] bf16, W:[N,K] bf16.
// Block = 128 threads = 4 waves; block tile 64(M) x 64(N); wave tile 16x64.
// Weight B-tiles are block-uniform -> staged in LDS by the TDM (double
// buffered, TENSORcnt synchronized), 4x less global weight traffic.
__global__ void __launch_bounds__(128)
cgemm_wmma(const unsigned short* __restrict__ Xr, const unsigned short* __restrict__ Xi,
           const unsigned short* __restrict__ Wr, const unsigned short* __restrict__ Wi,
           unsigned short* outRb, unsigned short* outIb,
           float* outRf, float* outIf,
           unsigned short* vTr, unsigned short* vTi,
           int M, int N, int K) {
  __shared__ unsigned short Wlds[2][2][64 * 32];  // [buf][re/im][n][k] 16 KB
  const int lane = threadIdx.x & 31;
  const int wid = threadIdx.x >> 5;
  const int hl = lane >> 4;  // half-wave
  const int m15 = lane & 15;
  const int rowBase = blockIdx.y * 64 + wid * 16;
  const int colBase = blockIdx.x * 64;
  const int row = rowBase + m15;
  const int nk = K / 32;

  const f32x8 fz = {0.f, 0.f, 0.f, 0.f, 0.f, 0.f, 0.f, 0.f};
  f32x8 accRn[4], accI[4];  // accRn holds NEGATED real accumulator
#pragma unroll
  for (int j = 0; j < 4; ++j) { accRn[j] = fz; accI[j] = fz; }

#ifdef HAVE_TDM
  if (wid == 0) {
    tdm_load_2d(lds_byte_off(&Wlds[0][0][0]), Wr + (size_t)colBase * K,
                (unsigned)K, (unsigned)N, 32u, 64u, (unsigned long long)K);
    tdm_load_2d(lds_byte_off(&Wlds[0][1][0]), Wi + (size_t)colBase * K,
                (unsigned)K, (unsigned)N, 32u, 64u, (unsigned long long)K);
  }
#endif

  for (int kc = 0; kc < nk; ++kc) {
    const int k0 = kc * 32;
    const int buf = kc & 1;
#ifdef HAVE_TDM
    if (wid == 0) {
      if (kc + 1 < nk) {
        const size_t goff = (size_t)colBase * K + (k0 + 32);
        tdm_load_2d(lds_byte_off(&Wlds[buf ^ 1][0][0]), Wr + goff, (unsigned)K,
                    (unsigned)N, 32u, 64u, (unsigned long long)K);
        tdm_load_2d(lds_byte_off(&Wlds[buf ^ 1][1][0]), Wi + goff, (unsigned)K,
                    (unsigned)N, 32u, 64u, (unsigned long long)K);
        __builtin_amdgcn_s_wait_tensorcnt(2);  // oldest pair done
      } else {
        __builtin_amdgcn_s_wait_tensorcnt(0);
      }
    }
    __syncthreads();
#else
    // cooperative fallback: 128 threads stage the two 64x32 tiles
    for (int idx = threadIdx.x; idx < 256; idx += 128) {
      const int n2 = idx >> 2;
      const int kk = (idx & 3) * 8;
      *(u32x4*)(&Wlds[buf][0][n2 * 32 + kk]) =
          *(const u32x4*)(Wr + (size_t)(colBase + n2) * K + k0 + kk);
      *(u32x4*)(&Wlds[buf][1][n2 * 32 + kk]) =
          *(const u32x4*)(Wi + (size_t)(colBase + n2) * K + k0 + kk);
    }
    __syncthreads();
#endif

    // A tiles (16x32): lane<16 -> K {0..7,16..23}, lane>=16 -> K {8..15,24..31}
    Tile ar, ai;
    const unsigned short* pxr = Xr + (size_t)row * K + k0 + hl * 8;
    ar.q[0] = *(const u32x4*)(pxr);
    ar.q[1] = *(const u32x4*)(pxr + 16);
    const unsigned short* pxi = Xi + (size_t)row * K + k0 + hl * 8;
    ai.q[0] = *(const u32x4*)(pxi);
    ai.q[1] = *(const u32x4*)(pxi + 16);
    __builtin_prefetch(pxr + 32, 0, 3);  // near-scope prefetch of next chunk
    __builtin_prefetch(pxi + 32, 0, 3);

    // Preload ALL 8 B tiles into distinct registers so the WMMA chain is not
    // gated on ds_cnt==0 between every pair (round-2 asm showed that bubble).
    const unsigned short* wr_t = &Wlds[buf][0][0];
    const unsigned short* wi_t = &Wlds[buf][1][0];
    Tile br[4], bi[4];
#pragma unroll
    for (int j = 0; j < 4; ++j) {
      const int n2 = j * 16 + m15;
      br[j].q[0] = *(const u32x4*)(wr_t + n2 * 32 + hl * 8);
      br[j].q[1] = *(const u32x4*)(wr_t + n2 * 32 + hl * 8 + 16);
      bi[j].q[0] = *(const u32x4*)(wi_t + n2 * 32 + hl * 8);
      bi[j].q[1] = *(const u32x4*)(wi_t + n2 * 32 + hl * 8 + 16);
    }
#pragma unroll
    for (int j = 0; j < 4; ++j) {
      // out_r += Xr*Wr - Xi*Wi  via negated accumulator:
      //   accRn' = Xi*Wi - (Xr*Wr - accRn)   (two C-negated WMMAs)
      accRn[j] = wmma_bf16_cn(ai, bi[j], wmma_bf16_cn(ar, br[j], accRn[j]));
      // out_i += Xr*Wi + Xi*Wr
      accI[j] = wmma_bf16(ar, bi[j], wmma_bf16(ai, br[j], accI[j]));
    }
    __syncthreads();
  }

  // C layout store: VGPR r, half hl -> row = rowBase + hl*8 + r, col = n
#pragma unroll
  for (int j = 0; j < 4; ++j) {
    const int ocol = colBase + j * 16 + m15;
#pragma unroll
    for (int r = 0; r < 8; ++r) {
      const int orow = rowBase + hl * 8 + r;
      const float vr = -accRn[j][r];  // undo negated accumulation
      const float vi = accI[j][r];
      const size_t idx = (size_t)orow * N + ocol;
      if (outRb) { outRb[idx] = f2bf(vr); outIb[idx] = f2bf(vi); }
      if (outRf) { outRf[idx] = vr; outIf[idx] = vi; }
      if (vTr && ocol >= 2 * DD) {  // transposed V: [B, DD, SS]
        const int bb = orow >> 10, s = orow & (SS - 1);
        const size_t tix = ((size_t)bb * DD + (ocol - 2 * DD)) * SS + s;
        vTr[tix] = f2bf(vr);
        vTi[tix] = f2bf(vi);
      }
    }
  }
}

// ---------------------------------------------------------------- attention
// One wave per 16-row Q tile; flash-style causal softmax over |Q conj(K)^T|.
// qkv: [B*S, 3*DD] bf16 (q at col 0, k at col DD); vT: [B, DD, SS] bf16.
// Output merged [B*S, DD] bf16.
__global__ void __launch_bounds__(128)
cattn_wmma(const unsigned short* __restrict__ qkvR, const unsigned short* __restrict__ qkvI,
           const unsigned short* __restrict__ vTr, const unsigned short* __restrict__ vTi,
           unsigned short* __restrict__ aoR, unsigned short* __restrict__ aoI) {
  __shared__ unsigned short Plds[4][16 * 32];  // per-wave P bounce buffer
  const int lane = threadIdx.x & 31;
  const int wid = threadIdx.x >> 5;
  const int hl = lane >> 4;
  const int m15 = lane & 15;
  const int bh = blockIdx.y;
  const int b = bh >> 4, h = bh & 15;
  const int q0 = (blockIdx.x * 4 + wid) * 16;
  const float scale = 0.125f;  // 1/sqrt(hd)
  const float NEGINF = -__builtin_inff();

  // Q tiles (A layout), two hd chunks of 32
  Tile qr[2], qi[2];
  {
    const size_t qrow = (size_t)(b * SS + q0 + m15) * (3 * DD) + h * HD;
#pragma unroll
    for (int c = 0; c < 2; ++c) {
      const unsigned short* pr = qkvR + qrow + c * 32 + hl * 8;
      qr[c].q[0] = *(const u32x4*)pr;
      qr[c].q[1] = *(const u32x4*)(pr + 16);
      const unsigned short* pi = qkvI + qrow + c * 32 + hl * 8;
      qi[c].q[0] = *(const u32x4*)pi;
      qi[c].q[1] = *(const u32x4*)(pi + 16);
    }
  }

  float mrun[8], lrun[8];
  const f32x8 fz = {0.f, 0.f, 0.f, 0.f, 0.f, 0.f, 0.f, 0.f};
  f32x8 accVr[4], accVi[4];
#pragma unroll
  for (int r = 0; r < 8; ++r) { mrun[r] = NEGINF; lrun[r] = 0.f; }
#pragma unroll
  for (int j = 0; j < 4; ++j) { accVr[j] = fz; accVi[j] = fz; }

  for (int kb = 0; kb <= q0 + 15; kb += 32) {
#pragma unroll
    for (int t = 0; t < 2; ++t) {
      const int keybase = kb + t * 16;
      // K^T B-tiles: lane holds key column (keybase+m15), contiguous hd.
      // All 4 tiles preloaded before the 8-WMMA score chain.
      Tile krb[2], kib[2];
      const size_t krow = (size_t)(b * SS + keybase + m15) * (3 * DD) + DD + h * HD;
#pragma unroll
      for (int c = 0; c < 2; ++c) {
        const unsigned short* pr = qkvR + krow + c * 32 + hl * 8;
        krb[c].q[0] = *(const u32x4*)pr;
        krb[c].q[1] = *(const u32x4*)(pr + 16);
        const unsigned short* pi = qkvI + krow + c * 32 + hl * 8;
        kib[c].q[0] = *(const u32x4*)pi;
        kib[c].q[1] = *(const u32x4*)(pi + 16);
      }
      // s_re = QrKr + QiKi
      f32x8 sre = fz;
      sre = wmma_bf16(qr[0], krb[0], sre);
      sre = wmma_bf16(qr[1], krb[1], sre);
      sre = wmma_bf16(qi[0], kib[0], sre);
      sre = wmma_bf16(qi[1], kib[1], sre);
      // s_im = QiKr - QrKi via one C-negated WMMA mid-chain:
      //   Qi1Kr1 + (Qi0Kr0 - (Qr1Ki1 + Qr0Ki0))
      f32x8 sim = fz;
      sim = wmma_bf16(qr[0], kib[0], sim);
      sim = wmma_bf16(qr[1], kib[1], sim);
      sim = wmma_bf16_cn(qi[0], krb[0], sim);
      sim = wmma_bf16(qi[1], krb[1], sim);

      const int coln = keybase + m15;
#pragma unroll
      for (int r = 0; r < 8; ++r) {
        const int rown = q0 + hl * 8 + r;
        const float re = sre[r], im = sim[r];
        float lg = sqrtf(re * re + im * im) * scale;
        lg = (coln <= rown) ? lg : NEGINF;
        // row max across the 16-lane half (row lives in one VGPR)
        float mx = lg;
        mx = fmaxf(mx, __shfl_xor(mx, 1, 32));
        mx = fmaxf(mx, __shfl_xor(mx, 2, 32));
        mx = fmaxf(mx, __shfl_xor(mx, 4, 32));
        mx = fmaxf(mx, __shfl_xor(mx, 8, 32));
        const float mnew = fmaxf(mrun[r], mx);
        float alpha, p;
        if (mnew == NEGINF) {  // fully-masked subtile
          alpha = 1.f;
          p = 0.f;
        } else {
          alpha = (mrun[r] == NEGINF) ? 0.f : __expf(mrun[r] - mnew);
          p = (lg == NEGINF) ? 0.f : __expf(lg - mnew);
        }
        float ps = p;
        ps += __shfl_xor(ps, 1, 32);
        ps += __shfl_xor(ps, 2, 32);
        ps += __shfl_xor(ps, 4, 32);
        ps += __shfl_xor(ps, 8, 32);
        lrun[r] = lrun[r] * alpha + ps;
        mrun[r] = mnew;
#pragma unroll
        for (int j = 0; j < 4; ++j) {
          accVr[j][r] *= alpha;
          accVi[j][r] *= alpha;
        }
        Plds[wid][(hl * 8 + r) * 32 + t * 16 + m15] = f2bf(p);
      }
    }
    // reload P (C layout -> A layout) through LDS; DS ops in-order per wave
    Tile pa;
    {
      const unsigned short* pp = &Plds[wid][m15 * 32 + hl * 8];
      pa.q[0] = *(const u32x4*)pp;
      pa.q[1] = *(const u32x4*)(pp + 16);
    }
    // out += P(16x32) @ V(32x64); V B-tiles contiguous thanks to vT layout.
    // Preload all 8 V tiles, then run the 8-WMMA chain unobstructed.
    Tile vbr[4], vbi[4];
#pragma unroll
    for (int j = 0; j < 4; ++j) {
      const size_t vrow =
          ((size_t)b * DD + h * HD + j * 16 + m15) * SS + kb + hl * 8;
      vbr[j].q[0] = *(const u32x4*)(vTr + vrow);
      vbr[j].q[1] = *(const u32x4*)(vTr + vrow + 16);
      vbi[j].q[0] = *(const u32x4*)(vTi + vrow);
      vbi[j].q[1] = *(const u32x4*)(vTi + vrow + 16);
    }
#pragma unroll
    for (int j = 0; j < 4; ++j) {
      accVr[j] = wmma_bf16(pa, vbr[j], accVr[j]);
      accVi[j] = wmma_bf16(pa, vbi[j], accVi[j]);
    }
  }

  // normalize and store merged [B*S, DD]
#pragma unroll
  for (int r = 0; r < 8; ++r) {
    const float inv = 1.0f / lrun[r];
    const int q = q0 + hl * 8 + r;
    const size_t rowoff = (size_t)(b * SS + q) * DD + h * HD;
#pragma unroll
    for (int j = 0; j < 4; ++j) {
      aoR[rowoff + j * 16 + m15] = f2bf(accVr[j][r] * inv);
      aoI[rowoff + j * 16 + m15] = f2bf(accVi[j][r] * inv);
    }
  }
}

// ---------------------------------------------------------------- launch
extern "C" void kernel_launch(void* const* d_in, const int* in_sizes, int n_in,
                              void* d_out, int out_size, void* d_ws, size_t ws_size,
                              hipStream_t stream) {
  const float* x_re = (const float*)d_in[0];
  const float* x_im = (const float*)d_in[1];
  const float* wqkv_re = (const float*)d_in[2];
  const float* wqkv_im = (const float*)d_in[3];
  const float* wproj_re = (const float*)d_in[4];
  const float* wproj_im = (const float*)d_in[5];

  unsigned char* base = (unsigned char*)d_ws;
  size_t off = 0;
  auto alloc_u16 = [&](size_t elems) {
    void* p = base + off;
    off += (elems * 2 + 255) & ~(size_t)255;
    return (unsigned short*)p;
  };
  unsigned short* xr = alloc_u16(BSD);           // x bf16
  unsigned short* xi = alloc_u16(BSD);
  unsigned short* wqr = alloc_u16(3 * DD * DD);  // wqkv bf16
  unsigned short* wqi = alloc_u16(3 * DD * DD);
  unsigned short* wpr = alloc_u16(DD * DD);      // wproj bf16
  unsigned short* wpi = alloc_u16(DD * DD);
  unsigned short* qkvR = alloc_u16((size_t)BB * SS * 3 * DD);  // qkv bf16
  unsigned short* qkvI = alloc_u16((size_t)BB * SS * 3 * DD);
  unsigned short* vTr = alloc_u16(BSD);          // V transposed [B,DD,SS]
  unsigned short* vTi = alloc_u16(BSD);
  unsigned short* aoR = alloc_u16(BSD);          // attn out merged bf16
  unsigned short* aoI = alloc_u16(BSD);

  // 1) fp32 -> bf16
  cvt_f32_bf16<<<2048, 256, 0, stream>>>(x_re, xr, BSD);
  cvt_f32_bf16<<<2048, 256, 0, stream>>>(x_im, xi, BSD);
  cvt_f32_bf16<<<2048, 256, 0, stream>>>(wqkv_re, wqr, 3 * DD * DD);
  cvt_f32_bf16<<<2048, 256, 0, stream>>>(wqkv_im, wqi, 3 * DD * DD);
  cvt_f32_bf16<<<1024, 256, 0, stream>>>(wproj_re, wpr, DD * DD);
  cvt_f32_bf16<<<1024, 256, 0, stream>>>(wproj_im, wpi, DD * DD);

  const dim3 blk(128);
  // 2) QKV complex GEMM: [4096,1024] x [3072,1024]^T, bf16 out + vT side store
  cgemm_wmma<<<dim3(3 * DD / 64, BB * SS / 64), blk, 0, stream>>>(
      xr, xi, wqr, wqi, qkvR, qkvI, nullptr, nullptr, vTr, vTi,
      BB * SS, 3 * DD, DD);

  // 3) fused causal complex attention (flash softmax over |scores|)
  cattn_wmma<<<dim3(SS / 64, BB * HH), blk, 0, stream>>>(qkvR, qkvI, vTr, vTi,
                                                         aoR, aoI);

  // 4) projection complex GEMM: fp32 out straight to d_out ([2,B,S,D])
  float* y_r = (float*)d_out;
  float* y_i = y_r + (size_t)BSD;
  cgemm_wmma<<<dim3(DD / 64, BB * SS / 64), blk, 0, stream>>>(
      aoR, aoI, wpr, wpi, nullptr, nullptr, y_r, y_i, nullptr, nullptr,
      BB * SS, DD, DD);
}
